// DRQ_19232863551819
// MI455X (gfx1250) — matrix operand order repaired and checked by reading, not compiled, not run on hardware
//
#include <hip/hip_runtime.h>

// CDNA5 / gfx1250 fused residual-quantizer, round 2.
// wave32; v_wmma_f32_16x16x32_bf16; LDS-staged bf16 codebook (one convert per
// WG per stage, fused ||c||^2); exact f32 residual update; write-BW bound.

typedef __attribute__((ext_vector_type(16))) __bf16 v16bf;
typedef __attribute__((ext_vector_type(8)))  __bf16 v8bf;
typedef __attribute__((ext_vector_type(8)))  float  v8f;
typedef __attribute__((ext_vector_type(4)))  float  f32x4;

#define RQ_N 65536
#define RQ_D 128
#define RQ_M 8
#define RQ_K 256
#define ROWS 64          // rows per workgroup (4 waves x 16 rows)
#define PITCH 132        // LDS residual row pitch (floats)
#define CBP 136          // LDS codebook row pitch (bf16): 272B = 17*16B
#define BLOCK 128        // 4 wave32

// dynamic LDS layout (bytes):
//   [0,               33792)   s_res    : ROWS*PITCH f32
//   [33792,          103424)   s_cb     : RQ_K*CBP bf16
//   [103424,         104448)   s_cnorm  : RQ_K f32
//   [104448,         104704)   s_argmin : ROWS int
#define SM_RES    0
#define SM_CB     33792
#define SM_CNORM  103424
#define SM_ARGMIN 104448
#define SM_BYTES  104704

__global__ __launch_bounds__(BLOCK) void rq_fused_kernel(
    const float* __restrict__ x,        // (N, D)
    const float* __restrict__ cb,       // (M, K, D) f32 codebooks
    float* __restrict__ out_xrec,       // (N, D)
    float* __restrict__ out_codes,      // (N, M, K) one-hot
    float* __restrict__ out_side)       // (M, N, D)
{
  extern __shared__ __align__(16) char smem[];
  float*  s_res    = (float*)(smem + SM_RES);
  __bf16* s_cb     = (__bf16*)(smem + SM_CB);
  float*  s_cnorm  = (float*)(smem + SM_CNORM);
  int*    s_argmin = (int*)(smem + SM_ARGMIN);

  const int tid     = threadIdx.x;
  const int lane    = tid & 31;
  const int wave    = tid >> 5;
  const int row0    = blockIdx.x * ROWS;
  const int rowbase = wave * 16;
  const int halfsel = lane >> 4;       // 0: lanes 0-15, 1: lanes 16-31
  const int lcol    = lane & 15;

  // ---- load x tile into LDS residual (res = x at stage 0) ----
  for (int i = tid; i < ROWS * (RQ_D / 4); i += BLOCK) {
    const int r = i >> 5, c4 = i & 31;
    const f32x4 v = *(const f32x4*)(x + (size_t)(row0 + r) * RQ_D + c4 * 4);
    *(f32x4*)(s_res + r * PITCH + c4 * 4) = v;
  }

  for (int s = 0; s < RQ_M; ++s) {
    // ---- stage codebook: f32 global -> bf16 LDS, fused ||c||^2 ----
    // thread handles one 8-float chunk; 16 consecutive threads share a row.
    {
      const float* cbs = cb + (size_t)s * RQ_K * RQ_D;
      for (int i = tid; i < RQ_K * (RQ_D / 8); i += BLOCK) {
        const int k = i >> 4, c8 = i & 15;
        const float* src = cbs + (size_t)k * RQ_D + c8 * 8;
        const f32x4 a = ((const f32x4*)src)[0];
        const f32x4 b = ((const f32x4*)src)[1];
        v8bf pk;
        pk[0] = (__bf16)a[0]; pk[1] = (__bf16)a[1];
        pk[2] = (__bf16)a[2]; pk[3] = (__bf16)a[3];
        pk[4] = (__bf16)b[0]; pk[5] = (__bf16)b[1];
        pk[6] = (__bf16)b[2]; pk[7] = (__bf16)b[3];
        *(v8bf*)(s_cb + k * CBP + c8 * 8) = pk;
        float nrm = a[0]*a[0] + a[1]*a[1] + a[2]*a[2] + a[3]*a[3]
                  + b[0]*b[0] + b[1]*b[1] + b[2]*b[2] + b[3]*b[3];
        for (int m = 1; m < 16; m <<= 1) nrm += __shfl_xor(nrm, m, 32);
        if ((tid & 15) == 0) s_cnorm[k] = nrm;
      }
    }
    // warm next stage's codebook toward L2 (global_prefetch_b8)
    if (s + 1 < RQ_M) {
      for (int pi = tid; pi < RQ_K; pi += BLOCK)
        __builtin_prefetch(cb + ((size_t)(s + 1) * RQ_K + pi) * RQ_D, 0, 0);
    }

    // ---- A fragments: (-2 * residual) bf16, contiguous K layout ----
    // lane m (<16): row m, K kb*32+0..15
    // lane m+16   : row m, K kb*32+16..31
    v16bf afrag[4];
    {
      const float* rp   = s_res + (rowbase + lcol) * PITCH;
      const int    base = halfsel * 16;
      for (int kb = 0; kb < 4; ++kb) {
        for (int p = 0; p < 8; ++p)
          afrag[kb][p]     = (__bf16)(-2.0f * rp[kb * 32 + base + p]);
        for (int p = 0; p < 8; ++p)
          afrag[kb][8 + p] = (__bf16)(-2.0f * rp[kb * 32 + base + 8 + p]);
      }
    }
    __syncthreads();   // staging + A built (A reads res, stable since last sync)

    float bestv[8];
    int   besti[8];
    for (int j = 0; j < 8; ++j) { bestv[j] = 3.4e38f; besti[j] = RQ_K; }

    // ---- 16 column tiles; 4 chained WMMAs each; B straight from LDS ----
    for (int t = 0; t < 16; ++t) {
      const int   col = t * 16 + lcol;
      const float cn  = s_cnorm[col];
      v8f acc = {cn, cn, cn, cn, cn, cn, cn, cn};     // seed with ||c||^2
      // B fragment: lane n (<16) = codeword n, K0..15 contiguous bf16;
      // lane n+16 = same codeword, K16..31.  Two ds_load_b128 per fragment.
      const __bf16* bp = s_cb + col * CBP + halfsel * 16;
      for (int kb = 0; kb < 4; ++kb) {
        const v16bf bfrag = *(const v16bf*)(bp + kb * 32);
        acc = __builtin_amdgcn_wmma_f32_16x16x32_bf16(
                  false, afrag[kb], false, bfrag, (short)0, acc, false, false);
      }
      for (int j = 0; j < 8; ++j) {   // VGPR j <-> row rowbase + halfsel*8 + j
        const float v = acc[j];
        if (v < bestv[j] || (v == bestv[j] && col < besti[j])) {
          bestv[j] = v; besti[j] = col;
        }
      }
    }

    // ---- cross-lane argmin within each 16-lane half (wave32 shuffles) ----
    for (int m = 1; m < 16; m <<= 1) {
      for (int j = 0; j < 8; ++j) {
        const float ov = __shfl_xor(bestv[j], m, 32);
        const int   oi = __shfl_xor(besti[j], m, 32);
        if (ov < bestv[j] || (ov == bestv[j] && oi < besti[j])) {
          bestv[j] = ov; besti[j] = oi;
        }
      }
    }
    if (lcol == 0) {
      for (int j = 0; j < 8; ++j)
        s_argmin[rowbase + halfsel * 8 + j] = besti[j];
    }
    __syncthreads();

    // ---- one-hot codes for this stage (coalesced 16B stores) ----
    for (int i = tid; i < ROWS * (RQ_K / 4); i += BLOCK) {
      const int r = i >> 6, q = i & 63;
      const int k = s_argmin[r];
      f32x4 v = {0.f, 0.f, 0.f, 0.f};
      if ((k >> 2) == q) ((float*)&v)[k & 3] = 1.0f;
      *(f32x4*)(out_codes + (size_t)(row0 + r) * (RQ_M * RQ_K)
                          + s * RQ_K + q * 4) = v;
    }
    // ---- exact f32 residual update + side_output[s] = x - res ----
    for (int i = tid; i < ROWS * (RQ_D / 4); i += BLOCK) {
      const int r = i >> 5, c4 = i & 31;
      const int k = s_argmin[r];
      const f32x4 cw = *(const f32x4*)(cb + ((size_t)(s * RQ_K + k)) * RQ_D + c4 * 4);
      f32x4 rv = *(f32x4*)(s_res + r * PITCH + c4 * 4);
      rv -= cw;
      *(f32x4*)(s_res + r * PITCH + c4 * 4) = rv;
      const f32x4 xv = *(const f32x4*)(x + (size_t)(row0 + r) * RQ_D + c4 * 4);
      const f32x4 xr = xv - rv;                     // x_recon after stage s
      *(f32x4*)(out_side + ((size_t)s * RQ_N + row0 + r) * RQ_D + c4 * 4) = xr;
      if (s == RQ_M - 1)
        *(f32x4*)(out_xrec + (size_t)(row0 + r) * RQ_D + c4 * 4) = xr;
    }
    __syncthreads();   // res final before next stage's A build / cb restage
  }
}

extern "C" void kernel_launch(void* const* d_in, const int* in_sizes, int n_in,
                              void* d_out, int out_size, void* d_ws, size_t ws_size,
                              hipStream_t stream) {
  (void)in_sizes; (void)n_in; (void)out_size; (void)d_ws; (void)ws_size;
  const float* x  = (const float*)d_in[0];   // (N, D)
  const float* cb = (const float*)d_in[1];   // (M, K, D)

  float* out = (float*)d_out;
  float* out_xrec  = out;                                        // N*D
  float* out_codes = out + (size_t)RQ_N * RQ_D;                  // N*M*K
  float* out_side  = out + (size_t)RQ_N * RQ_D
                         + (size_t)RQ_N * RQ_M * RQ_K;           // M*N*D

  rq_fused_kernel<<<RQ_N / ROWS, BLOCK, SM_BYTES, stream>>>(x, cb, out_xrec,
                                                            out_codes, out_side);
}
